// SparseSAKELayer_48387101556868
// MI455X (gfx1250) — compile-verified, matrix operand order checked by reading, not run on hardware
//
#include <hip/hip_runtime.h>
#include <hip/hip_bf16.h>
#include <math.h>

// ---------------------------------------------------------------------------
// SAKE layer for MI455X (gfx1250, wave32, WMMA bf16 16x16x32, TDM gather)
// ---------------------------------------------------------------------------

#define NN   50000
#define EE   800000
#define NK   50      // RBF kernels (padded to 64 for WMMA)
#define GAMMA_ 10.0f
#define EPS_  1e-5f

typedef __attribute__((ext_vector_type(16))) __bf16 v16bf;
typedef __attribute__((ext_vector_type(8)))  float  v8f;
typedef __attribute__((ext_vector_type(4)))  int    v4i;
typedef __attribute__((ext_vector_type(8)))  int    v8i;

#define WMMA_BF16(A, B, C) \
  __builtin_amdgcn_wmma_f32_16x16x32_bf16(false, (A), false, (B), (short)0, (C), false, false)

__device__ __forceinline__ float silu_f(float v) { return v / (1.0f + __expf(-v)); }

// ---- fragment-layout helpers (ISA 7.12.2, 16-bit 16x32 A / 32x16 B, wave32) ----
__device__ __forceinline__ int swz(int m, int k) {
  const int kk = k & 31;
  return (((k >> 5) * 32 + m + ((kk & 8) << 1)) << 4) + ((kk & 16) >> 1) + (kk & 7);
}
__device__ __forceinline__ v16bf ld_a_swz(const __bf16* tile, int kt, int lane) {
  return *(const v16bf*)(tile + ((kt * 32 + lane) << 4));
}
__device__ __forceinline__ v16bf ld_a_f32(const float* t, int k0, int lane) {
  v16bf a;
  const int m  = lane & 15;
  const int kh = (lane & 16) >> 1;
#pragma unroll
  for (int j = 0; j < 16; ++j) {
    const int k = k0 + ((j & 8) << 1) + kh + (j & 7);
    a[j] = (__bf16)t[m * 128 + k];
  }
  return a;
}
__device__ __forceinline__ v16bf ld_b_frag(const __bf16* wf, int fbase, int kt,
                                           int NT, int nt, int lane) {
  return *(const v16bf*)(wf + (((fbase + kt * NT + nt) * 32 + lane) << 4));
}

// ---- Tensor Data Mover gather (CDNA5 D# gather mode, TENSORcnt) ----
#define RFL(x) __builtin_amdgcn_readfirstlane((int)(x))
__device__ __forceinline__ void tdm_gather(v4i g0, v8i g1, v4i g2, v4i g3) {
  asm volatile("tensor_load_to_lds %0, %1, %2, %3"
               :: "s"(g0), "s"(g1), "s"(g2), "s"(g3) : "memory");
}
__device__ __forceinline__ int pack16(int a, int b) {
  return (a & 0xFFFF) | (b << 16);
}

// ---- workspace layout (floats); h_e stored as bf16 ----
#define OFF_HEB     ((size_t)0)                         // [E,64] bf16 = E*32 floats
#define OFF_EEU     ((size_t)EE * 32)                   // [E]
#define OFF_ESEM    (OFF_EEU  + (size_t)EE)             // [E]
#define OFF_ET      (OFF_ESEM + (size_t)EE)             // [E]
#define OFF_SUMEU   (OFF_ET   + (size_t)EE)             // [N]
#define OFF_SUMSEM  (OFF_SUMEU  + (size_t)NN)           // [N]
#define OFF_SUMT    (OFF_SUMSEM + (size_t)NN)           // [N]
#define OFF_XATT    (OFF_SUMT   + (size_t)NN)           // [N,32,3]
#define OFF_HAGG    (OFF_XATT + (size_t)NN * 96)        // [N,64]
#define OFF_XDEL    (OFF_HAGG + (size_t)NN * 64)        // [N,3]
#define OFF_WF      (OFF_XDEL + (size_t)NN * 3)         // 88 frags * 512 bf16
#define ZERO_CNT    ((size_t)NN * 166)                  // SUMEU..XDEL contiguous

// fragment bases (units of 512-bf16 fragments)
#define FB_CFIN 0
#define FB_CFF  16
#define FB_EW   24
#define FB_C1   36
#define FB_PN1  44
#define FB_PN2  48
#define FB_N1   56
#define FB_N2   80
#define NFRAG   88

// ---------------------------------------------------------------------------
__global__ void sake_zero(float* p, size_t n) {
  size_t i = (size_t)blockIdx.x * blockDim.x + threadIdx.x;
  if (i < n) p[i] = 0.0f;
}

// ---------------------------------------------------------------------------
__global__ void sake_swz_w(
    const float* __restrict__ W_cf_in, const float* __restrict__ W_cf_f,
    const float* __restrict__ W_ew,    const float* __restrict__ W_c1,
    const float* __restrict__ W_pn1,   const float* __restrict__ W_pn2,
    const float* __restrict__ W_n1,    const float* __restrict__ W_n2,
    __bf16* __restrict__ wf) {
  const int tid = blockIdx.x * blockDim.x + threadIdx.x;
  if (tid >= NFRAG * 512) return;
  const int f = tid >> 9, r = tid & 511;
  const int lane = r >> 4, j = r & 15;
  const int kl = ((j & 8) << 1) + ((lane & 16) >> 1) + (j & 7);
  const int nl = lane & 15;

  const float* W; int f0, ldn, kmax, NT;
  if      (f < FB_CFF) { W = W_cf_in; f0 = f;           ldn = 64; kmax = 128; NT = 4; }
  else if (f < FB_EW)  { W = W_cf_f;  f0 = f - FB_CFF;  ldn = 64; kmax = NK;  NT = 4; }
  else if (f < FB_C1)  { W = W_ew;    f0 = f - FB_EW;   ldn = 32; kmax = 192; NT = 2; }
  else if (f < FB_PN1) { W = W_c1;    f0 = f - FB_C1;   ldn = 64; kmax = 64;  NT = 4; }
  else if (f < FB_PN2) { W = W_pn1;   f0 = f - FB_PN1;  ldn = 64; kmax = 32;  NT = 4; }
  else if (f < FB_N1)  { W = W_pn2;   f0 = f - FB_PN2;  ldn = 64; kmax = 64;  NT = 4; }
  else if (f < FB_N2)  { W = W_n1;    f0 = f - FB_N1;   ldn = 64; kmax = 192; NT = 4; }
  else                 { W = W_n2;    f0 = f - FB_N2;   ldn = 64; kmax = 64;  NT = 4; }
  const int kt = f0 / NT, nt = f0 % NT;
  const int k = kt * 32 + kl, n = nt * 16 + nl;
  wf[tid] = (__bf16)((k < kmax) ? W[k * ldn + n] : 0.0f);
}

// ---------------------------------------------------------------------------
// Edge pass A: one wave32 per 16-edge tile; grid is exact (no guards needed).
// ---------------------------------------------------------------------------
#define WAVES 4
__global__ __launch_bounds__(128) void sake_edge_a(
    const float* __restrict__ h, const float* __restrict__ x,
    const int* __restrict__ src, const int* __restrict__ dst,
    const float* __restrict__ b_cf_in, const float* __restrict__ b_cf_f,
    const float* __restrict__ W_sem,   const float* __restrict__ b_ew,
    const float* __restrict__ b_c1,
    const float* __restrict__ W_c2,    const float* __restrict__ b_c2,
    const __bf16* __restrict__ wf, float* __restrict__ ws) {
  __shared__ __align__(32) float  s_hcf[WAVES][16 * 128];  // fp32 h_cat (TDM target)
  __shared__ __align__(32) __bf16 s_aux[WAVES][16 * 64];   // swizzled rbf / silu(c1)
  __shared__ __align__(32) __bf16 s_heb[WAVES][16 * 64];   // swizzled h_e
  __shared__ int    s_src[WAVES][16];
  __shared__ int    s_dst[WAVES][16];
  __shared__ float  s_xmx[WAVES][16 * 3];
  __shared__ float  s_inv[WAVES][16];
  __shared__ float  s_d  [WAVES][16];

  const int wid  = threadIdx.x >> 5;          // wave in block (warpSize==32)
  const int lane = threadIdx.x & 31;
  const int tile = blockIdx.x * WAVES + wid;
  const int e0   = tile * 16;

  float*  hcf = s_hcf[wid];
  __bf16* aux = s_aux[wid];
  __bf16* heb = s_heb[wid];

  __bf16* ws_heb = (__bf16*)ws;               // h_e (bf16) at workspace start
  float* ws_eeu  = ws + OFF_EEU;
  float* ws_esem = ws + OFF_ESEM;
  float* sum_eu  = ws + OFF_SUMEU;
  float* sum_sem = ws + OFF_SUMSEM;
  float* xatt    = ws + OFF_XATT;
  float* xdel    = ws + OFF_XDEL;

  // ---- phase 1: geometry + softmax(eu) denominator ----
  if (lane < 16) {
    const int e = e0 + lane;
    const int s = src[e], t = dst[e];
    s_src[wid][lane] = s;
    s_dst[wid][lane] = t;
    float d2 = 0.0f;
#pragma unroll
    for (int dim = 0; dim < 3; ++dim) {
      const float mx = x[s * 3 + dim] - x[t * 3 + dim];
      s_xmx[wid][lane * 3 + dim] = mx;
      d2 += mx * mx;
    }
    const float d = sqrtf(fmaxf(d2, 0.0f) + EPS_);
    s_d  [wid][lane] = d;
    s_inv[wid][lane] = 1.0f / (d * d);
    const float eeu = __expf(-d);
    ws_eeu[e] = eeu;
    atomicAdd(&sum_eu[t], eeu);
  }
  __syncthreads();

  // ---- TDM gather: two gather-mode descriptors (src rows / dst rows) ----
  // rows of 64 f32 (256B), LDS padding 64 DW every 64 DW -> 512B row stride,
  // interleaving src half (offset 0) and dst half (offset 256B).
  {
    const unsigned hcbase = (unsigned)(size_t)&hcf[0];
    const unsigned long long ga = (unsigned long long)(size_t)h;
    v8i g1;
    g1[0] = (2 << 16) | (1 << 20) | (5 << 22) | (63 << 25); // 4B, pad 64DW/64DW
    g1[1] = (64 << 16);                 // tensor_dim0 = 64
    g1[2] = (int)((NN & 0xFFFF) << 16); // tensor_dim1 lo
    g1[3] = (64 << 16) | (NN >> 16);    // tile_dim0 = 64, tensor_dim1 hi
    g1[4] = 16;                         // tile_dim1 = 16 valid indices
    g1[5] = 64;                         // tensor_dim0_stride = 64
    g1[6] = 0; g1[7] = 0;
    const int galo = RFL((unsigned)ga);
    const int gahi = (RFL((unsigned)(ga >> 32)) & 0x01FFFFFF) | (2 << 30);

    v4i i0s, i1s, i0d, i1d;
#pragma unroll
    for (int q = 0; q < 4; ++q) {
      i0s[q] = RFL(pack16(s_src[wid][2 * q],     s_src[wid][2 * q + 1]));
      i1s[q] = RFL(pack16(s_src[wid][2 * q + 8], s_src[wid][2 * q + 9]));
      i0d[q] = RFL(pack16(s_dst[wid][2 * q],     s_dst[wid][2 * q + 1]));
      i1d[q] = RFL(pack16(s_dst[wid][2 * q + 8], s_dst[wid][2 * q + 9]));
    }
    v4i g0;
    g0[0] = (int)0x80000001u;           // count=1, gather_mode=1, 16-bit idx
    g0[2] = galo; g0[3] = gahi;
    g0[1] = RFL(hcbase);
    tdm_gather(g0, g1, i0s, i1s);       // src rows -> hcf + m*512
    g0[1] = RFL(hcbase + 256);
    tdm_gather(g0, g1, i0d, i1d);       // dst rows -> hcf + 256 + m*512
  }

  // ---- RBF tile (swizzled bf16) overlapped with the TDM copies ----
  {
    const int m  = lane & 15;
    const int kb = (lane & 16) * 2;
    const float d = s_d[wid][m];
#pragma unroll
    for (int kk = 0; kk < 32; ++kk) {
      const int k = kb + kk;
      float v = 0.0f;
      if (k < NK) {
        const float mu = 5.0f * (float)k / 49.0f;
        v = __expf(-GAMMA_ * (d - mu) * (d - mu));
      }
      aux[swz(m, k)] = (__bf16)v;
    }
  }
  __builtin_amdgcn_s_wait_tensorcnt(0);
  __syncthreads();

  const int n16  = lane & 15;
  const int mrow = (lane & 16) >> 1;

  // ---- phase 2: h_e = (h_cat@W_cf_in + b) * silu(rbf@W_cf_f + b) ----
  v16bf a_hc[4];                              // cached for reuse in phase 3
#pragma unroll
  for (int kt = 0; kt < 4; ++kt) a_hc[kt] = ld_a_f32(hcf, kt * 32, lane);
  {
    v8f acc[4] = {}, fac[4] = {};             // inline-0 C for first WMMA
#pragma unroll
    for (int kt = 0; kt < 4; ++kt)
#pragma unroll
      for (int nt = 0; nt < 4; ++nt)
        acc[nt] = WMMA_BF16(a_hc[kt], ld_b_frag(wf, FB_CFIN, kt, 4, nt, lane), acc[nt]);
#pragma unroll
    for (int kt = 0; kt < 2; ++kt) {
      const v16bf a = ld_a_swz(aux, kt, lane);
#pragma unroll
      for (int nt = 0; nt < 4; ++nt)
        fac[nt] = WMMA_BF16(a, ld_b_frag(wf, FB_CFF, kt, 4, nt, lane), fac[nt]);
    }
#pragma unroll
    for (int nt = 0; nt < 4; ++nt) {
      const float ba = b_cf_in[nt * 16 + n16];
      const float bf = b_cf_f [nt * 16 + n16];
#pragma unroll
      for (int r = 0; r < 8; ++r) {
        const float he = (acc[nt][r] + ba) * silu_f(fac[nt][r] + bf);
        const int m = r + mrow, n = nt * 16 + n16;
        ws_heb[(size_t)(e0 + m) * 64 + n] = (__bf16)he;
        heb[swz(m, n)] = (__bf16)he;
      }
    }
    if (lane < 16) {
      float sdot = 0.0f;
#pragma unroll 16
      for (int k = 0; k < 128; ++k) sdot += hcf[lane * 128 + k] * W_sem[k];
      const float es = __expf(silu_f(sdot));
      ws_esem[e0 + lane] = es;
      atomicAdd(&sum_sem[s_dst[wid][lane]], es);
    }
  }
  __syncthreads();

  // ---- phase 3: how = tanh([h_cat|h_e]@W_ew+b); x_att scatter; coef GEMM1 ----
  v16bf a_heb[2];
#pragma unroll
  for (int kt = 0; kt < 2; ++kt) a_heb[kt] = ld_a_swz(heb, kt, lane);
  {
    v8f hw[2] = {};
#pragma unroll
    for (int kt = 0; kt < 6; ++kt) {
      const v16bf a = (kt < 4) ? a_hc[kt] : a_heb[kt - 4];
#pragma unroll
      for (int nt = 0; nt < 2; ++nt)
        hw[nt] = WMMA_BF16(a, ld_b_frag(wf, FB_EW, kt, 2, nt, lane), hw[nt]);
    }
#pragma unroll
    for (int nt = 0; nt < 2; ++nt) {
      const float bb = b_ew[nt * 16 + n16];
#pragma unroll
      for (int r = 0; r < 8; ++r) {
        const float howv = tanhf(hw[nt][r] + bb);
        const int m = r + mrow, c = nt * 16 + n16;
        const int t = s_dst[wid][m];
        const float inv = s_inv[wid][m];
#pragma unroll
        for (int dim = 0; dim < 3; ++dim)
          atomicAdd(&xatt[(size_t)t * 96 + c * 3 + dim],
                    howv * s_xmx[wid][m * 3 + dim] * inv);
      }
    }
    v8f cc[4] = {};
#pragma unroll
    for (int kt = 0; kt < 2; ++kt)
#pragma unroll
      for (int nt = 0; nt < 4; ++nt)
        cc[nt] = WMMA_BF16(a_heb[kt], ld_b_frag(wf, FB_C1, kt, 4, nt, lane), cc[nt]);
#pragma unroll
    for (int nt = 0; nt < 4; ++nt) {
      const float bb = b_c1[nt * 16 + n16];
#pragma unroll
      for (int r = 0; r < 8; ++r)
        aux[swz(r + mrow, nt * 16 + n16)] = (__bf16)silu_f(cc[nt][r] + bb);
    }
  }
  __syncthreads();

  // ---- phase 4: coef = aux @ W_c2 + b_c2 ; x_delta scatter ----
  if (lane < 16) {
    float s = b_c2[0];
#pragma unroll 16
    for (int k = 0; k < 64; ++k) s += (float)aux[swz(lane, k)] * W_c2[k];
    const int t = s_dst[wid][lane];
#pragma unroll
    for (int dim = 0; dim < 3; ++dim)
      atomicAdd(&xdel[(size_t)t * 3 + dim], s_xmx[wid][lane * 3 + dim] * s);
  }
}

// ---------------------------------------------------------------------------
__global__ void sake_edge_b(const int* __restrict__ dst, float* __restrict__ ws) {
  const int e = blockIdx.x * blockDim.x + threadIdx.x;
  if (e >= EE) return;
  const int t = dst[e];
  const float eu  = ws[OFF_EEU  + e] / ws[OFF_SUMEU  + t];
  const float sem = ws[OFF_ESEM + e] / ws[OFF_SUMSEM + t];
  const float et  = __expf(eu * sem);
  ws[OFF_ET + e] = et;
  atomicAdd(&ws[OFF_SUMT + t], et);
}

// ---------------------------------------------------------------------------
__global__ void sake_edge_c(const int* __restrict__ dst, float* __restrict__ ws) {
  const long long tid = (long long)blockIdx.x * blockDim.x + threadIdx.x;
  if (tid >= (long long)EE * 16) return;
  const int e  = (int)(tid >> 4);
  const int j0 = ((int)tid & 15) * 4;
  const int t  = dst[e];
  const float tot = ws[OFF_ET + e] / ws[OFF_SUMT + t];
  const __bf16* he = (const __bf16*)ws + (size_t)e * 64 + j0;
  float* ha = ws + OFF_HAGG + (size_t)t * 64 + j0;
#pragma unroll
  for (int j = 0; j < 4; ++j) atomicAdd(&ha[j], tot * (float)he[j]);
}

// ---------------------------------------------------------------------------
__global__ __launch_bounds__(128) void sake_node(
    const float* __restrict__ h, const float* __restrict__ x,
    const float* __restrict__ b_pn1, const float* __restrict__ b_pn2,
    const float* __restrict__ b_n1,  const float* __restrict__ b_n2,
    const __bf16* __restrict__ wf,
    const float* __restrict__ ws, float* __restrict__ out) {
  __shared__ __align__(32) __bf16 s_cat[WAVES][16 * 192];
  __shared__ __align__(32) __bf16 s_t64[WAVES][16 * 64];
  __shared__ __align__(32) __bf16 s_nrm[WAVES][16 * 32];

  const int wid  = threadIdx.x >> 5;
  const int lane = threadIdx.x & 31;
  const int tile = blockIdx.x * WAVES + wid;
  const bool ok  = tile < (NN / 16);
  const int  v0  = tile * 16;

  __bf16* cat = s_cat[wid];
  __bf16* t64 = s_t64[wid];
  __bf16* nrm = s_nrm[wid];

  const float* xatt = ws + OFF_XATT;
  const float* hagg = ws + OFF_HAGG;
  const float* xdel = ws + OFF_XDEL;

  const int n16  = lane & 15;
  const int mrow = (lane & 16) >> 1;

  if (ok) {
    const int m  = lane & 15;
    const int cb = lane & 16;
#pragma unroll
    for (int ci = 0; ci < 16; ++ci) {
      const int c = cb + ci;
      const float* p = xatt + (size_t)(v0 + m) * 96 + c * 3;
      const float s = p[0] * p[0] + p[1] * p[1] + p[2] * p[2];
      nrm[swz(m, c)] = (__bf16)sqrtf(fmaxf(s, 0.0f) + EPS_);
    }
    for (int idx = lane; idx < 16 * 128; idx += 32) {
      const int mm = idx >> 7, k = idx & 127;
      const float v = (k < 64) ? h[(size_t)(v0 + mm) * 64 + k]
                               : hagg[(size_t)(v0 + mm) * 64 + (k - 64)];
      cat[swz(mm, k)] = (__bf16)v;
    }
  }
  __syncthreads();

  if (ok) {
    v8f p[4] = {};
    const v16bf a = ld_a_swz(nrm, 0, lane);
#pragma unroll
    for (int nt = 0; nt < 4; ++nt)
      p[nt] = WMMA_BF16(a, ld_b_frag(wf, FB_PN1, 0, 4, nt, lane), p[nt]);
#pragma unroll
    for (int nt = 0; nt < 4; ++nt) {
      const float bb = b_pn1[nt * 16 + n16];
#pragma unroll
      for (int r = 0; r < 8; ++r)
        t64[swz(r + mrow, nt * 16 + n16)] = (__bf16)silu_f(p[nt][r] + bb);
    }
  }
  __syncthreads();

  if (ok) {
    v8f q[4] = {};
#pragma unroll
    for (int kt = 0; kt < 2; ++kt) {
      const v16bf a = ld_a_swz(t64, kt, lane);
#pragma unroll
      for (int nt = 0; nt < 4; ++nt)
        q[nt] = WMMA_BF16(a, ld_b_frag(wf, FB_PN2, kt, 4, nt, lane), q[nt]);
    }
#pragma unroll
    for (int nt = 0; nt < 4; ++nt) {
      const float bb = b_pn2[nt * 16 + n16];
#pragma unroll
      for (int r = 0; r < 8; ++r)
        cat[swz(r + mrow, 128 + nt * 16 + n16)] = (__bf16)(q[nt][r] + bb);
    }
  }
  __syncthreads();

  if (ok) {
    v8f p[4] = {};
#pragma unroll
    for (int kt = 0; kt < 6; ++kt) {
      const v16bf a = ld_a_swz(cat, kt, lane);
#pragma unroll
      for (int nt = 0; nt < 4; ++nt)
        p[nt] = WMMA_BF16(a, ld_b_frag(wf, FB_N1, kt, 4, nt, lane), p[nt]);
    }
#pragma unroll
    for (int nt = 0; nt < 4; ++nt) {
      const float bb = b_n1[nt * 16 + n16];
#pragma unroll
      for (int r = 0; r < 8; ++r)
        t64[swz(r + mrow, nt * 16 + n16)] = (__bf16)silu_f(p[nt][r] + bb);
    }
  }
  __syncthreads();

  if (ok) {
    v8f q[4] = {};
#pragma unroll
    for (int kt = 0; kt < 2; ++kt) {
      const v16bf a = ld_a_swz(t64, kt, lane);
#pragma unroll
      for (int nt = 0; nt < 4; ++nt)
        q[nt] = WMMA_BF16(a, ld_b_frag(wf, FB_N2, kt, 4, nt, lane), q[nt]);
    }
#pragma unroll
    for (int nt = 0; nt < 4; ++nt) {
      const float bb = b_n2[nt * 16 + n16];
#pragma unroll
      for (int r = 0; r < 8; ++r)
        out[(size_t)(v0 + r + mrow) * 64 + nt * 16 + n16] = q[nt][r] + bb;
    }
    if (lane < 16) {
      const int v = v0 + lane;
#pragma unroll
      for (int dim = 0; dim < 3; ++dim)
        out[(size_t)NN * 64 + (size_t)v * 3 + dim] =
            x[(size_t)v * 3 + dim] + xdel[(size_t)v * 3 + dim];
    }
  }
}

// ---------------------------------------------------------------------------
extern "C" void kernel_launch(void* const* d_in, const int* in_sizes, int n_in,
                              void* d_out, int out_size, void* d_ws, size_t ws_size,
                              hipStream_t stream) {
  const float* h       = (const float*)d_in[0];
  const float* x       = (const float*)d_in[1];
  const int*   src     = (const int*)  d_in[2];
  const int*   dst     = (const int*)  d_in[3];
  const float* W_cf_in = (const float*)d_in[4];
  const float* b_cf_in = (const float*)d_in[5];
  const float* W_cf_f  = (const float*)d_in[6];
  const float* b_cf_f  = (const float*)d_in[7];
  const float* W_sem   = (const float*)d_in[8];
  const float* W_ew    = (const float*)d_in[9];
  const float* b_ew    = (const float*)d_in[10];
  const float* W_pn1   = (const float*)d_in[11];
  const float* b_pn1   = (const float*)d_in[12];
  const float* W_pn2   = (const float*)d_in[13];
  const float* b_pn2   = (const float*)d_in[14];
  const float* W_n1    = (const float*)d_in[15];
  const float* b_n1    = (const float*)d_in[16];
  const float* W_n2    = (const float*)d_in[17];
  const float* b_n2    = (const float*)d_in[18];
  const float* W_c1    = (const float*)d_in[19];
  const float* b_c1    = (const float*)d_in[20];
  const float* W_c2    = (const float*)d_in[21];
  const float* b_c2    = (const float*)d_in[22];
  float*  ws  = (float*)d_ws;
  __bf16* wf  = (__bf16*)(ws + OFF_WF);
  float*  out = (float*)d_out;

  sake_zero<<<(unsigned)((ZERO_CNT + 255) / 256), 256, 0, stream>>>(ws + OFF_SUMEU, ZERO_CNT);
  sake_swz_w<<<(NFRAG * 512 + 255) / 256, 256, 0, stream>>>(
      W_cf_in, W_cf_f, W_ew, W_c1, W_pn1, W_pn2, W_n1, W_n2, wf);
  sake_edge_a<<<EE / 16 / WAVES, 128, 0, stream>>>(
      h, x, src, dst, b_cf_in, b_cf_f, W_sem, b_ew, b_c1, W_c2, b_c2, wf, ws);
  sake_edge_b<<<(EE + 255) / 256, 256, 0, stream>>>(dst, ws);
  sake_edge_c<<<(EE * 16) / 256, 256, 0, stream>>>(dst, ws);
  sake_node<<<(NN / 16 + WAVES - 1) / WAVES, 128, 0, stream>>>(
      h, x, b_pn1, b_pn2, b_n1, b_n2, wf, ws, out);
}
